// FleetModel_8538394985162
// MI455X (gfx1250) — matrix operand order repaired and checked by reading, not compiled, not run on hardware
//
#include <hip/hip_runtime.h>
#include <hip/hip_bf16.h>

typedef __attribute__((ext_vector_type(16))) _Float16 v16h;
typedef __attribute__((ext_vector_type(8)))  float    v8f;
typedef _Float16 h16;

#define NTOK    160
#define DMODEL  32
#define NHEAD   4
#define HDIM    8
#define FFDIM   128
#define PVTOK   40
#define NTHREADS 320   // 10 waves of 32; wave w owns rows [16w,16w+16)

struct FleetParams {
  const float *states, *lane_ctx, *mask, *pos_emb;
  const float *pos_w, *pos_b, *hdg_w, *hdg_b, *lan_w, *lan_b;
  const float *traj_w, *traj_b, *int_w, *int_b;
  const float *in_w[2], *in_b[2], *out_w[2], *out_b[2];
  const float *ln1_g[2], *ln1_b[2], *w1[2], *b1[2], *w2[2], *b2[2];
  const float *ln2_g[2], *ln2_b[2];
};

// D = A(16x32 f16) * B(32x16 f16) + C(f32)
__device__ inline v8f wmma_f16(v16h a, v16h b, v8f c) {
  return __builtin_amdgcn_wmma_f32_16x16x32_f16(false, a, false, b, (short)0, c,
                                                false, false);
}

// A-fragment (16x32 f16) from LDS, row-major buf with given stride (halfs).
// Lane l<16: row m0+l, K slots k0..k0+15 contiguous;
// lane l>=16: row m0+l-16, K slots k0+16..k0+31 contiguous.
__device__ inline v16h frag_a_lds(const h16* buf, int stride, int m0, int k0, int lane) {
  int r  = m0 + (lane & 15);
  int kb = k0 + ((lane & 16) ? 16 : 0);
  const h16* p = buf + r * stride + kb;
  v16h f;
#pragma unroll
  for (int i = 0; i < 16; ++i) f[i] = p[i];
  return f;
}

// B-fragment (32x16) where B[k][n] = W[n][k], W row-major (Nout x K) f32 in global.
// Lane l: n = n0 + (l&15); K slots k0 + (l<16 ? 0..15 : 16..31) contiguous.
__device__ inline v16h frag_b_wt(const float* W, int K, int n0, int k0, int lane) {
  int n  = n0 + (lane & 15);
  int kb = k0 + ((lane & 16) ? 16 : 0);
  const float* p = W + n * K + kb;
  v16h f;
#pragma unroll
  for (int i = 0; i < 16; ++i) f[i] = (h16)p[i];
  return f;
}

__device__ inline void layernorm_row(const float* src, float* dstC, h16* dstH,
                                     const float* g, const float* beta, int row) {
  const float* s = src + row * DMODEL;
  float mean = 0.f;
#pragma unroll
  for (int d = 0; d < DMODEL; ++d) mean += s[d];
  mean *= (1.0f / DMODEL);
  float var = 0.f;
#pragma unroll
  for (int d = 0; d < DMODEL; ++d) { float t = s[d] - mean; var += t * t; }
  var *= (1.0f / DMODEL);
  float inv = rsqrtf(var + 1e-5f);
#pragma unroll
  for (int d = 0; d < DMODEL; ++d) {
    float y = (s[d] - mean) * inv * g[d] + beta[d];
    dstC[row * DMODEL + d] = y;
    dstH[row * DMODEL + d] = (h16)y;
  }
}

__global__ __launch_bounds__(NTHREADS)
void fleet_kernel(FleetParams P, float* __restrict__ out, int B) {
  // LDS carve (123520 B total):
  //   [0)      sC   f32 160x32 canvas            (20480)
  //   [20480)  sCh  f16 160x32 canvas mirror     (10240)
  //   [30720)  sRS  f32 160 softmax 1/sum        (  640)
  //   [31360)  sQKV f16 160x96                   (30720)   alias: sT f32 160x32 / sH f16 160x128
  //   [62080)  sAO  f16 160x32 attn out          (10240)   (covered by sH alias, dead by then)
  //   [72320)  sS   f16 160x160 scores           (51200)   alias: sT2 f32 160x32
  __shared__ __align__(16) char smem[123520];
  float* sC   = (float*)smem;
  h16*   sCh  = (h16*)(smem + 20480);
  float* sRS  = (float*)(smem + 30720);
  h16*   sQKV = (h16*)(smem + 31360);
  h16*   sAO  = (h16*)(smem + 62080);
  h16*   sS   = (h16*)(smem + 72320);
  float* sT   = (float*)(smem + 31360);
  h16*   sH   = (h16*)(smem + 31360);
  float* sT2  = (float*)(smem + 72320);

  const int tid  = threadIdx.x;
  const int lane = tid & 31;
  const int m0   = (tid >> 5) * 16;   // wave's row tile
  const int b    = blockIdx.x;

  // ---- canvas init: pos_emb + scattered pos/hdg/lane projections (VALU) ----
  if (tid < NTOK) {
    const int n = tid, v = n / PVTOK, off = n % PVTOK;
    const float* st = P.states   + ((size_t)b * 4 + v) * 4;
    const float* lc = P.lane_ctx + ((size_t)b * 4 + v) * 4;
    const float s0 = st[0], s1 = st[1], s2 = st[2], s3 = st[3];
    const float l0 = lc[0], l1 = lc[1], l2 = lc[2], l3 = lc[3];
    const float sp = fmaxf(sqrtf(s2 * s2 + s3 * s3), 0.1f);
    const float h0 = s3 / sp, h1 = s2 / sp;
#pragma unroll 4
    for (int d = 0; d < DMODEL; ++d) {
      float acc = P.pos_emb[n * DMODEL + d];
      if (off < 8) {
        int r = off * DMODEL + d;
        acc += s0 * P.pos_w[r*4+0] + s1 * P.pos_w[r*4+1] +
               s2 * P.pos_w[r*4+2] + s3 * P.pos_w[r*4+3] + P.pos_b[r];
      } else if (off < 12) {
        int r = (off - 8) * DMODEL + d;
        acc += h0 * P.hdg_w[r*2+0] + h1 * P.hdg_w[r*2+1] + P.hdg_b[r];
      } else if (off < 20) {
        int r = (off - 12) * DMODEL + d;
        acc += l0 * P.lan_w[r*4+0] + l1 * P.lan_w[r*4+1] +
               l2 * P.lan_w[r*4+2] + l3 * P.lan_w[r*4+3] + P.lan_b[r];
      }
      sC[n * DMODEL + d]  = acc;
      sCh[n * DMODEL + d] = (h16)acc;
    }
  }
  __syncthreads();

  for (int L = 0; L < 2; ++L) {
    // ---- QKV: (160x32)x(32x96), 6 N-tiles per wave ----
    for (int jt = 0; jt < 6; ++jt) {
      v16h a  = frag_a_lds(sCh, DMODEL, m0, 0, lane);
      v16h bb = frag_b_wt(P.in_w[L], DMODEL, jt * 16, 0, lane);
      v8f  c  = {};
      c = wmma_f16(a, bb, c);
      int n  = jt * 16 + (lane & 15);
      int mb = m0 + ((lane & 16) ? 8 : 0);
      float bias = P.in_b[L][n];
#pragma unroll
      for (int r = 0; r < 8; ++r)
        sQKV[(mb + r) * 96 + n] = (h16)(c[r] + bias);
    }
    __syncthreads();

    // ---- attention, one head at a time ----
    for (int h = 0; h < NHEAD; ++h) {
      // S = Q K^T * 1/sqrt(8); K padded 8->32 with zeros (exact)
      for (int kt = 0; kt < 10; ++kt) {
        v16h a = {}, bb = {};
        if (lane < 16) {
          int qrow = m0 + lane;
          int key  = kt * 16 + lane;
#pragma unroll
          for (int i = 0; i < 8; ++i) {
            a[i]  = sQKV[qrow * 96 + h * HDIM + i];          // Q dims in K-slots 0..7
            bb[i] = sQKV[key * 96 + 32 + h * HDIM + i];      // K dims in K-slots 0..7
          }
        }
        v8f c = {};
        c = wmma_f16(a, bb, c);
        int n  = kt * 16 + (lane & 15);
        int mb = m0 + ((lane & 16) ? 8 : 0);
#pragma unroll
        for (int r = 0; r < 8; ++r)
          sS[(mb + r) * NTOK + n] = (h16)(c[r] * 0.35355339059f);
      }
      __syncthreads();

      // row softmax (unnormalized exp; keep reciprocal rowsum)
      if (tid < NTOK) {
        const float* mrow = P.mask + tid * NTOK;
        h16* srow = sS + tid * NTOK;
        float mx = -3.0e38f;
        for (int k = 0; k < NTOK; ++k)
          mx = fmaxf(mx, (float)srow[k] + mrow[k]);
        float sum = 0.f;
        for (int k = 0; k < NTOK; ++k) {
          float p = __expf(((float)srow[k] + mrow[k]) - mx);
          srow[k] = (h16)p;
          sum += p;
        }
        sRS[tid] = 1.0f / sum;
      }
      __syncthreads();

      // O = P V, K=160 in 5 chained steps; V in N-cols 0..7
      {
        v8f c = {};
        for (int kt = 0; kt < 5; ++kt) {
          v16h a  = frag_a_lds(sS, NTOK, m0, kt * 32, lane);
          v16h bb = {};
          int n = lane & 15;
          if (n < HDIM) {
            int kb = kt * 32 + ((lane & 16) ? 16 : 0);
#pragma unroll
            for (int i = 0; i < 16; ++i)
              bb[i] = sQKV[(kb + i) * 96 + 64 + h * HDIM + n];
          }
          c = wmma_f16(a, bb, c);
        }
        int n  = lane & 15;
        int mb = m0 + ((lane & 16) ? 8 : 0);
        if (n < HDIM) {
#pragma unroll
          for (int r = 0; r < 8; ++r)
            sAO[(mb + r) * DMODEL + h * HDIM + n] = (h16)(c[r] * sRS[mb + r]);
        }
      }
      __syncthreads();
    }

    // ---- out projection + residual ----
    for (int jt = 0; jt < 2; ++jt) {
      v16h a  = frag_a_lds(sAO, DMODEL, m0, 0, lane);
      v16h bb = frag_b_wt(P.out_w[L], DMODEL, jt * 16, 0, lane);
      v8f  c  = {};
      c = wmma_f16(a, bb, c);
      int n  = jt * 16 + (lane & 15);
      int mb = m0 + ((lane & 16) ? 8 : 0);
      float bias = P.out_b[L][n];
#pragma unroll
      for (int r = 0; r < 8; ++r) {
        int row = mb + r;
        sT[row * DMODEL + n] = c[r] + bias + sC[row * DMODEL + n];
      }
    }
    __syncthreads();
    if (tid < NTOK) layernorm_row(sT, sC, sCh, P.ln1_g[L], P.ln1_b[L], tid);
    __syncthreads();

    // ---- FFN1: (160x32)x(32x128) + ReLU ----
    for (int jt = 0; jt < 8; ++jt) {
      v16h a  = frag_a_lds(sCh, DMODEL, m0, 0, lane);
      v16h bb = frag_b_wt(P.w1[L], DMODEL, jt * 16, 0, lane);
      v8f  c  = {};
      c = wmma_f16(a, bb, c);
      int n  = jt * 16 + (lane & 15);
      int mb = m0 + ((lane & 16) ? 8 : 0);
      float bias = P.b1[L][n];
#pragma unroll
      for (int r = 0; r < 8; ++r)
        sH[(mb + r) * FFDIM + n] = (h16)fmaxf(c[r] + bias, 0.f);
    }
    __syncthreads();

    // ---- FFN2: (160x128)x(128x32) + residual ----
    for (int jt = 0; jt < 2; ++jt) {
      v8f c = {};
      for (int kt = 0; kt < 4; ++kt) {
        v16h a  = frag_a_lds(sH, FFDIM, m0, kt * 32, lane);
        v16h bb = frag_b_wt(P.w2[L], FFDIM, jt * 16, kt * 32, lane);
        c = wmma_f16(a, bb, c);
      }
      int n  = jt * 16 + (lane & 15);
      int mb = m0 + ((lane & 16) ? 8 : 0);
      float bias = P.b2[L][n];
#pragma unroll
      for (int r = 0; r < 8; ++r) {
        int row = mb + r;
        sT2[row * DMODEL + n] = c[r] + bias + sC[row * DMODEL + n];
      }
    }
    __syncthreads();
    if (tid < NTOK) layernorm_row(sT2, sC, sCh, P.ln2_g[L], P.ln2_b[L], tid);
    __syncthreads();
  }

  // ---- output heads (tiny GEMVs, VALU; wave-uniform branches) ----
  if (tid < 64) {                       // trajs: (B,V,PH,2) flat = b*64 + v*16 + j
    int v = tid >> 4, j = tid & 15;
    float acc = P.traj_b[j];
    const float* w = P.traj_w + j * 512;
    int tok0 = v * PVTOK + 20;
    for (int q = 0; q < 512; ++q)
      acc += sC[(tok0 + (q >> 5)) * DMODEL + (q & 31)] * w[q];
    out[(size_t)b * 64 + tid] = acc;
  } else if (tid < 96) {                // intents at offset B*64: b*32 + v*8 + j
    int t = tid - 64, v = t >> 3, j = t & 7;
    float acc = P.int_b[j];
    const float* w = P.int_w + j * 128;
    int tok0 = v * PVTOK + 36;
    for (int q = 0; q < 128; ++q)
      acc += sC[(tok0 + (q >> 5)) * DMODEL + (q & 31)] * w[q];
    out[(size_t)B * 64 + (size_t)b * 32 + t] = acc;
  }
}

extern "C" void kernel_launch(void* const* d_in, const int* in_sizes, int n_in,
                              void* d_out, int out_size, void* d_ws, size_t ws_size,
                              hipStream_t stream) {
  (void)n_in; (void)out_size; (void)d_ws; (void)ws_size;
  FleetParams P;
  P.states   = (const float*)d_in[0];
  P.lane_ctx = (const float*)d_in[1];
  P.mask     = (const float*)d_in[2];
  P.pos_emb  = (const float*)d_in[3];
  P.pos_w  = (const float*)d_in[4];  P.pos_b  = (const float*)d_in[5];
  P.hdg_w  = (const float*)d_in[6];  P.hdg_b  = (const float*)d_in[7];
  P.lan_w  = (const float*)d_in[8];  P.lan_b  = (const float*)d_in[9];
  P.traj_w = (const float*)d_in[10]; P.traj_b = (const float*)d_in[11];
  P.int_w  = (const float*)d_in[12]; P.int_b  = (const float*)d_in[13];
  for (int L = 0; L < 2; ++L) {
    const int o = 14 + L * 12;
    P.in_w[L]  = (const float*)d_in[o + 0];  P.in_b[L]  = (const float*)d_in[o + 1];
    P.out_w[L] = (const float*)d_in[o + 2];  P.out_b[L] = (const float*)d_in[o + 3];
    P.ln1_g[L] = (const float*)d_in[o + 4];  P.ln1_b[L] = (const float*)d_in[o + 5];
    P.w1[L]    = (const float*)d_in[o + 6];  P.b1[L]    = (const float*)d_in[o + 7];
    P.w2[L]    = (const float*)d_in[o + 8];  P.b2[L]    = (const float*)d_in[o + 9];
    P.ln2_g[L] = (const float*)d_in[o + 10]; P.ln2_b[L] = (const float*)d_in[o + 11];
  }
  const int B = in_sizes[0] / 16;           // states is (B, V=4, 4)
  fleet_kernel<<<dim3(B), dim3(NTHREADS), 0, stream>>>(P, (float*)d_out, B);
}
